// DeltaNetLayer_9079560864063
// MI455X (gfx1250) — matrix operand order repaired and checked by reading, not compiled, not run on hardware
//
#include <hip/hip_runtime.h>

// ---------------------------------------------------------------------------
// DeltaNet layer for MI455X (gfx1250, wave32, WMMA 16x16x32 bf16, TDM staging)
// Pipeline:
//  1) k_ln        : LayerNorm -> bf16 normed activations, + beta = sigmoid(n@Wb)
//  2) k_tr        : weight convert f32->bf16 + transpose (B-operand friendly)
//  3) k_gemm_qkv  : [8192x1024]@[1024x4096] bf16 WMMA GEMM, SiLU epilogue
//  4) k_l2n       : per-head l2norm of q,k (q also * DK^-0.5) -> bf16
//  5) k_scan      : chunked delta rule, state S^T slice in LDS, WMMA GEMMs
//  6) k_rms       : per-head RMSNorm of o -> bf16
//  7) k_gemm_out  : [8192x2048]@[2048x1024] bf16 WMMA GEMM, +x residual
// GEMM tiles are staged LDS-side by the Tensor Data Mover (tensor_load_to_lds,
// double buffered, s_wait_tensorcnt pipelining) when the builtin is available.
// ---------------------------------------------------------------------------

typedef __attribute__((ext_vector_type(16))) __bf16 v16bf;
typedef __attribute__((ext_vector_type(8)))  __bf16 v8bf;
typedef __attribute__((ext_vector_type(8)))  float  v8f;

#define DMODEL 1024
#define NH     4
#define DKH    256
#define DVH    512
#define NB     4
#define NT     2048
#define NTOK   (NB*NT)   // 8192
#define CHUNK  32        // chunk length for chunked delta rule
#define DVS    32        // dv slice per workgroup

// ---------------- Tensor Data Mover staging (CDNA5 ISA ch. 8) --------------

#if defined(__has_builtin)
#  if __has_builtin(__builtin_amdgcn_tensor_load_to_lds) && \
      __has_builtin(__builtin_amdgcn_s_wait_tensorcnt)
#    define USE_TDM 1
#  endif
#endif

#ifdef USE_TDM
typedef __attribute__((ext_vector_type(4))) unsigned int v4u;
typedef __attribute__((ext_vector_type(8))) int          v8i;
typedef __attribute__((ext_vector_type(4))) int          v4i;

// Load a [rows x cols] bf16 tile (row stride = `stride` elements in global,
// tight row-major in LDS) via one TDM descriptor.  D# bitfields per ISA §8.3/8.4.
// 6-arg builtin form (amdgpu-toolchain clang-23):
//   (uint32x4 g0, int32x8 g1, int32x4 g2, int32x4 g3, int32x8 g4, i32 cpol)
__device__ __forceinline__ void tdm_load_2d_bf16(const __bf16* gptr, void* lds_ptr,
                                                 unsigned rows, unsigned cols,
                                                 unsigned stride, unsigned trows) {
  unsigned long long ga = (unsigned long long)(size_t)gptr;
  unsigned lds = (unsigned)(size_t)lds_ptr;
  v4u g0;
  g0.x = 1u;                                        // count=1, user descriptor
  g0.y = lds;                                       // lds_addr (bytes)
  g0.z = (unsigned)(ga & 0xffffffffu);              // global_addr[31:0]
  g0.w = ((unsigned)(ga >> 32) & 0x01ffffffu)       // global_addr[56:32]
         | (2u << 30);                              // type = 2 ("image")
  unsigned td0 = stride;                            // tensor dim0 (elements)
  unsigned td1 = trows;                             // tensor dim1 (rows)
  v8i g1;
  g1[0] = (int)(1u << 16);                          // data_size = 1 (2 bytes)
  g1[1] = (int)((td0 & 0xffffu) << 16);             // tensor_dim0[15:0]
  g1[2] = (int)((td0 >> 16) | ((td1 & 0xffffu) << 16));
  g1[3] = (int)((td1 >> 16) | (cols << 16));        // tile_dim0 = cols
  g1[4] = (int)rows;                                // tile_dim1 = rows, tile_dim2=0
  g1[5] = (int)stride;                              // tensor_dim0_stride[31:0]
  g1[6] = 0;                                        // stride[47:32] | dim1_stride lo
  g1[7] = 0;
  v4i z4 = {0, 0, 0, 0};
  v8i z8 = {0, 0, 0, 0, 0, 0, 0, 0};
  __builtin_amdgcn_tensor_load_to_lds(g0, g1, z4, z4, z8, 0);
}
#endif

// ---------------- WMMA helpers (layouts per CDNA5 ISA 7.12.2) ---------------

__device__ __forceinline__ v8f wmma_bf(v16bf a, v16bf b, v8f c) {
  return __builtin_amdgcn_wmma_f32_16x16x32_bf16(false, a, false, b, (short)0, c,
                                                 false, false);
}

__device__ __forceinline__ v8f zero8() {
  v8f z = {0.f,0.f,0.f,0.f,0.f,0.f,0.f,0.f};
  return z;
}

// A operand (16x32 bf16) from a row-major bf16 tile; base points at row m0.
__device__ __forceinline__ v16bf ld_a_bf(const __bf16* base, int ld, int kk, int lane) {
  int m  = lane & 15;
  int kb = kk + ((lane < 16) ? 0 : 8);
  const __bf16* p = base + m * ld + kb;
  union { v16bf v; v8bf h[2]; } u;
  u.h[0] = *(const v8bf*)(p);
  u.h[1] = *(const v8bf*)(p + 16);
  return u.v;
}

// A operand from a row-major f32 tile (convert to bf16 in regs).
__device__ __forceinline__ v16bf ld_a_f32(const float* base, int ld, int kk, int lane) {
  int m  = lane & 15;
  int kb = kk + ((lane < 16) ? 0 : 8);
  const float* p = base + m * ld + kb;
  v16bf r;
#pragma unroll
  for (int i = 0; i < 8; ++i) r[i] = (__bf16)p[i];
#pragma unroll
  for (int i = 0; i < 8; ++i) r[8 + i] = (__bf16)p[16 + i];
  return r;
}

// B operand (32x16 bf16) from LDS holding B^T row-major (rows = N, cols = K).
__device__ __forceinline__ v16bf ld_b_bt(const __bf16* base, int ld, int kk, int lane) {
  int n  = lane & 15;
  int kb = kk + ((lane < 16) ? 0 : 8);
  const __bf16* p = base + n * ld + kb;
  union { v16bf v; v8bf h[2]; } u;
  u.h[0] = *(const v8bf*)(p);
  u.h[1] = *(const v8bf*)(p + 16);
  return u.v;
}

// B operand from a *row-major B* tile (rows = K contraction, cols = N): gather.
__device__ __forceinline__ v16bf ld_b_rm(const __bf16* base, int ld, int lane) {
  int n  = lane & 15;
  int kb = (lane < 16) ? 0 : 8;
  v16bf r;
#pragma unroll
  for (int i = 0; i < 16; ++i) r[i] = base[(kb + (i < 8 ? i : i + 8)) * ld + n];
  return r;
}

__device__ __forceinline__ float sigmoidf_(float x) { return 1.f / (1.f + __expf(-x)); }
__device__ __forceinline__ float siluf_(float x)    { return x * sigmoidf_(x); }

// ---------------- 1) LayerNorm + beta -------------------------------------

__global__ __launch_bounds__(256) void k_ln(const float* __restrict__ x,
                                            const float* __restrict__ lnw,
                                            const float* __restrict__ lnb,
                                            const float* __restrict__ Wb,
                                            __bf16* __restrict__ normed,
                                            float* __restrict__ betab) {
  __shared__ float red[256];
  int t = blockIdx.x, tid = threadIdx.x;
  const float* xr = x + (size_t)t * DMODEL;
  float v[4]; float s = 0.f;
#pragma unroll
  for (int i = 0; i < 4; ++i) { v[i] = xr[tid * 4 + i]; s += v[i]; }
  red[tid] = s; __syncthreads();
  for (int st = 128; st; st >>= 1) { if (tid < st) red[tid] += red[tid + st]; __syncthreads(); }
  float mu = red[0] * (1.f / DMODEL); __syncthreads();
  float s2 = 0.f;
#pragma unroll
  for (int i = 0; i < 4; ++i) { float d = v[i] - mu; s2 += d * d; }
  red[tid] = s2; __syncthreads();
  for (int st = 128; st; st >>= 1) { if (tid < st) red[tid] += red[tid + st]; __syncthreads(); }
  float inv = rsqrtf(red[0] * (1.f / DMODEL) + 1e-5f); __syncthreads();

  float bacc[NH] = {0.f, 0.f, 0.f, 0.f};
#pragma unroll
  for (int i = 0; i < 4; ++i) {
    int d = tid * 4 + i;
    float nv = (v[i] - mu) * inv * lnw[d] + lnb[d];
    normed[(size_t)t * DMODEL + d] = (__bf16)nv;
#pragma unroll
    for (int h = 0; h < NH; ++h) bacc[h] += nv * Wb[(size_t)d * NH + h];
  }
  for (int h = 0; h < NH; ++h) {
    red[tid] = bacc[h]; __syncthreads();
    for (int st = 128; st; st >>= 1) { if (tid < st) red[tid] += red[tid + st]; __syncthreads(); }
    if (tid == 0) betab[(size_t)t * NH + h] = sigmoidf_(red[0]);
    __syncthreads();
  }
}

// ---------------- 2) Weight convert + transpose ----------------------------

__global__ __launch_bounds__(256) void k_tr(const float* __restrict__ src,
                                            __bf16* __restrict__ dst,
                                            int R, int Ccol) {
  size_t i = (size_t)blockIdx.x * 256 + threadIdx.x;
  if (i >= (size_t)R * Ccol) return;
  size_t c = i / R, r = i % R;
  dst[i] = (__bf16)src[r * (size_t)Ccol + c];
}

// ---------------- GEMM core (tile 128x128, K-step 32, double buffered) ----
// A row-major bf16 [M][K], Bt = B^T row-major bf16 [N][K].  Staged by TDM.

#define GEMM_BODY(KDIM, MTOT, NTOT, EPILOGUE)                                     \
  __shared__ __bf16 As[2][128][32];                                               \
  __shared__ __bf16 Bs[2][128][32];                                               \
  int tid = threadIdx.x, lane = tid & 31, w = tid >> 5;                           \
  int m0 = blockIdx.x * 128, n0 = blockIdx.y * 128;                               \
  int wr = (w & 1) * 64, wc = (w >> 1) * 32;                                      \
  v8f acc[4][2];                                                                  \
  _Pragma("unroll") for (int i = 0; i < 4; ++i)                                   \
    _Pragma("unroll") for (int j = 0; j < 2; ++j) acc[i][j] = zero8();            \
  int lr = tid >> 1, lc = (tid & 1) * 16;                                         \
  const __bf16* agbase = A  + (size_t)(m0 + lr) * (KDIM) + lc;                    \
  const __bf16* bgbase = Bt + (size_t)(n0 + lr) * (KDIM) + lc;                    \
  (void)agbase; (void)bgbase;                                                     \
  STAGE(0, 0);                                                                    \
  const int NSTEP = (KDIM) / 32;                                                  \
  for (int s = 0; s < NSTEP; ++s) {                                               \
    int cur = s & 1;                                                              \
    if (s + 1 < NSTEP) { STAGE(cur ^ 1, (s + 1) * 32); TWAIT(2); }                \
    else               { TWAIT(0); }                                              \
    __syncthreads();                                                              \
    v16bf b0 = ld_b_bt(&Bs[cur][wc][0],      32, 0, lane);                        \
    v16bf b1 = ld_b_bt(&Bs[cur][wc + 16][0], 32, 0, lane);                        \
    _Pragma("unroll") for (int i = 0; i < 4; ++i) {                               \
      v16bf a = ld_a_bf(&As[cur][wr + i * 16][0], 32, 0, lane);                   \
      acc[i][0] = wmma_bf(a, b0, acc[i][0]);                                      \
      acc[i][1] = wmma_bf(a, b1, acc[i][1]);                                      \
    }                                                                             \
    __syncthreads();                                                              \
  }                                                                               \
  int mh = (lane < 16) ? 0 : 8, nl = lane & 15;                                   \
  _Pragma("unroll") for (int i = 0; i < 4; ++i)                                   \
    _Pragma("unroll") for (int j = 0; j < 2; ++j)                                 \
      _Pragma("unroll") for (int r = 0; r < 8; ++r) {                             \
        int mg = m0 + wr + i * 16 + r + mh;                                       \
        int ng = n0 + wc + j * 16 + nl;                                           \
        EPILOGUE;                                                                 \
      }

#ifdef USE_TDM
#define STAGE(buf, kk)                                                            \
  do { if (w == 0) {                                                              \
    tdm_load_2d_bf16(A  + (size_t)m0 * KD_ + (kk), &As[buf][0][0], 128, 32, KD_, MT_); \
    tdm_load_2d_bf16(Bt + (size_t)n0 * KD_ + (kk), &Bs[buf][0][0], 128, 32, KD_, NT_); \
  } } while (0)
#define TWAIT(n) do { if (w == 0) __builtin_amdgcn_s_wait_tensorcnt(n); } while (0)
#else
#define STAGE(buf, kk)                                                            \
  do {                                                                            \
    *(uint4*)&As[buf][lr][lc]     = *(const uint4*)(agbase + (kk));               \
    *(uint4*)&As[buf][lr][lc + 8] = *(const uint4*)(agbase + (kk) + 8);           \
    *(uint4*)&Bs[buf][lr][lc]     = *(const uint4*)(bgbase + (kk));               \
    *(uint4*)&Bs[buf][lr][lc + 8] = *(const uint4*)(bgbase + (kk) + 8);           \
    __builtin_prefetch(agbase + (kk) + 32, 0, 1);                                 \
    __builtin_prefetch(bgbase + (kk) + 32, 0, 1);                                 \
  } while (0)
#define TWAIT(n) do { } while (0)
#endif

// ---------------- 3) QKV GEMM: normed[8192x1024] @ Wqkv[1024x4096] ---------

__global__ __launch_bounds__(256) void k_gemm_qkv(const __bf16* __restrict__ A,
                                                  const __bf16* __restrict__ Bt,
                                                  float* __restrict__ qk,
                                                  __bf16* __restrict__ vbuf) {
  enum { KD_ = DMODEL, MT_ = NTOK, NT_ = 4096 };
  GEMM_BODY(DMODEL, NTOK, 4096, {
    float val = siluf_(acc[i][j][r]);
    if (ng < 2048) qk[(size_t)mg * 2048 + ng] = val;
    else           vbuf[(size_t)mg * 2048 + (ng - 2048)] = (__bf16)val;
  })
}

// ---------------- 7) Output GEMM: ob[8192x2048] @ Wo[2048x1024] + x --------

__global__ __launch_bounds__(256) void k_gemm_out(const __bf16* __restrict__ A,
                                                  const __bf16* __restrict__ Bt,
                                                  const float* __restrict__ x,
                                                  float* __restrict__ out) {
  enum { KD_ = 2048, MT_ = NTOK, NT_ = DMODEL };
  GEMM_BODY(2048, NTOK, DMODEL, {
    size_t idx = (size_t)mg * DMODEL + ng;
    out[idx] = x[idx] + acc[i][j][r];
  })
}

// ---------------- 4) l2norm of q,k per (token, head) -----------------------

__global__ __launch_bounds__(256) void k_l2n(const float* __restrict__ qk,
                                             __bf16* __restrict__ qb,
                                             __bf16* __restrict__ kb) {
  __shared__ float red[256];
  int t = blockIdx.x, tid = threadIdx.x;
  const float* row = qk + (size_t)t * 2048;
  float v[8]; float s = 0.f;
#pragma unroll
  for (int i = 0; i < 8; ++i) { v[i] = row[tid * 8 + i]; s += v[i] * v[i]; }
  red[tid] = s; __syncthreads();
  int g = tid >> 5, l = tid & 31;   // 8 groups of 32 threads: one 256-wide head each
  for (int st = 16; st; st >>= 1) { if (l < st) red[tid] += red[tid + st]; __syncthreads(); }
  float inv = rsqrtf(red[g << 5] + 1e-6f);
  if (g < 4) inv *= 0.0625f;        // q heads also * DK^-0.5 = 1/16
#pragma unroll
  for (int i = 0; i < 8; ++i) {
    int col = tid * 8 + i;
    float nv = v[i] * inv;
    if (col < 1024) qb[(size_t)t * 1024 + col]          = (__bf16)nv;
    else            kb[(size_t)t * 1024 + (col - 1024)] = (__bf16)nv;
  }
}

// ---------------- 5) Chunked delta-rule scan -------------------------------
// grid = (DV/DVS = 16 slices, B*H = 16), 256 threads (8 waves).
// State slice S^T [DVS x DK] bf16 persists in LDS for all T/CHUNK chunks.

__global__ __launch_bounds__(256) void k_scan(const __bf16* __restrict__ qb,
                                              const __bf16* __restrict__ kb,
                                              const __bf16* __restrict__ vb,
                                              const float* __restrict__ betab,
                                              float* __restrict__ o) {
  __shared__ __bf16 Sbt[DVS][DKH];     // state^T  [v][k]        16 KB
  __shared__ __bf16 Kc[CHUNK][DKH];    // K chunk  [c][k]        16 KB
  __shared__ __bf16 Qc[CHUNK][DKH];    // Q chunk  [c][k]        16 KB
  __shared__ float  Aatt[CHUNK][CHUNK];// tril(b K K^T)           4 KB
  __shared__ __bf16 Mm[CHUNK][CHUNK];  // tril(Q K^T) incl diag   2 KB
  __shared__ float  VDt[DVS][CHUNK];   // VD^T  [v][c]            4 KB
  __shared__ float  Vt[DVS][CHUNK];    // V^T   [v][c]            4 KB
  __shared__ float  betac[CHUNK];

  int slice = blockIdx.x;              // 0..15
  int bh    = blockIdx.y;              // 0..15
  int b = bh / NH, h = bh % NH;
  int tid = threadIdx.x, lane = tid & 31, w = tid >> 5;
  int mh = (lane < 16) ? 0 : 8, nl = lane & 15;

  for (int i = tid; i < DVS * DKH; i += 256) ((__bf16*)Sbt)[i] = (__bf16)0.f;
  __syncthreads();

  const int tokbase = b * NT;
  const int qkcol   = h * DKH;
  const int vcol    = h * DVH + slice * DVS;

  for (int ch = 0; ch < NT / CHUNK; ++ch) {
    int t0 = ch * CHUNK;
    // ---- stage K,Q chunk (TDM when available), V^T slice, beta ----
    {
#ifdef USE_TDM
      if (w == 0) {
        tdm_load_2d_bf16(kb + (size_t)(tokbase + t0) * 1024 + qkcol, &Kc[0][0],
                         CHUNK, DKH, 1024, NTOK);
        tdm_load_2d_bf16(qb + (size_t)(tokbase + t0) * 1024 + qkcol, &Qc[0][0],
                         CHUNK, DKH, 1024, NTOK);
      }
#else
      int r = tid >> 3, cc = (tid & 7) * 32;
      const __bf16* kg = kb + (size_t)(tokbase + t0 + r) * 1024 + qkcol + cc;
      const __bf16* qg = qb + (size_t)(tokbase + t0 + r) * 1024 + qkcol + cc;
#pragma unroll
      for (int u = 0; u < 4; ++u) {
        *(uint4*)&Kc[r][cc + u * 8] = *(const uint4*)(kg + u * 8);
        *(uint4*)&Qc[r][cc + u * 8] = *(const uint4*)(qg + u * 8);
      }
#endif
      int c2 = tid >> 3, v0 = (tid & 7) * 4;
      const __bf16* vg = vb + (size_t)(tokbase + t0 + c2) * 2048 + vcol + v0;
#pragma unroll
      for (int u = 0; u < 4; ++u) Vt[v0 + u][c2] = (float)vg[u];
      if (tid < CHUNK) betac[tid] = betab[(size_t)(tokbase + t0 + tid) * NH + h];
#ifdef USE_TDM
      if (w == 0) __builtin_amdgcn_s_wait_tensorcnt(0);
#endif
    }
    __syncthreads();

    // ---- phase 1: waves 0-3: RHS^T = S^T K^T ; waves 4-7: Aatt = tril(b KK^T)
    if (w < 4) {
      int iv = (w >> 1) * 16, ic = (w & 1) * 16;
      v8f acc = zero8();
      for (int kk = 0; kk < DKH; kk += 32) {
        v16bf a = ld_a_bf(&Sbt[iv][0], DKH, kk, lane);
        v16bf bb = ld_b_bt(&Kc[ic][0], DKH, kk, lane);
        acc = wmma_bf(a, bb, acc);
      }
#pragma unroll
      for (int r = 0; r < 8; ++r) {
        int v = iv + r + mh, c = ic + nl;
        VDt[v][c] = betac[c] * (Vt[v][c] - acc[r]);   // raw RHS, solved below
      }
    } else {
      int i0 = ((w - 4) >> 1) * 16, j0 = ((w - 4) & 1) * 16;
      v8f acc = zero8();
      for (int kk = 0; kk < DKH; kk += 32) {
        v16bf a = ld_a_bf(&Kc[i0][0], DKH, kk, lane);
        v16bf bb = ld_b_bt(&Kc[j0][0], DKH, kk, lane);
        acc = wmma_bf(a, bb, acc);
      }
#pragma unroll
      for (int r = 0; r < 8; ++r) {
        int i = i0 + r + mh, j = j0 + nl;
        Aatt[i][j] = (j < i) ? betac[i] * acc[r] : 0.f;
      }
    }
    __syncthreads();

    // ---- phase 2: waves 0-3: O1^T = S^T Q^T ; waves 4-7: M = tril(Q K^T)
    v8f oacc = zero8();
    int iv = 0, ic = 0;
    if (w < 4) {
      iv = (w >> 1) * 16; ic = (w & 1) * 16;
      for (int kk = 0; kk < DKH; kk += 32) {
        v16bf a = ld_a_bf(&Sbt[iv][0], DKH, kk, lane);
        v16bf bb = ld_b_bt(&Qc[ic][0], DKH, kk, lane);
        oacc = wmma_bf(a, bb, oacc);
      }
    } else {
      int i0 = ((w - 4) >> 1) * 16, j0 = ((w - 4) & 1) * 16;
      v8f acc = zero8();
      for (int kk = 0; kk < DKH; kk += 32) {
        v16bf a = ld_a_bf(&Qc[i0][0], DKH, kk, lane);
        v16bf bb = ld_b_bt(&Kc[j0][0], DKH, kk, lane);
        acc = wmma_bf(a, bb, acc);
      }
#pragma unroll
      for (int r = 0; r < 8; ++r) {
        int i = i0 + r + mh, j = j0 + nl;
        Mm[i][j] = (__bf16)((j <= i) ? acc[r] : 0.f);
      }
    }
    __syncthreads();

    // ---- forward substitution: VD = (I + Aatt)^-1 RHS, per v-column ----
    if (tid < DVS) {
      int v = tid;
      for (int i = 1; i < CHUNK; ++i) {
        float a = VDt[v][i];
        for (int j = 0; j < i; ++j) a -= Aatt[i][j] * VDt[v][j];
        VDt[v][i] = a;
      }
    }
    __syncthreads();

    // ---- O^T = O1^T + VD^T M^T ; write o ----
    if (w < 4) {
      v16bf a = ld_a_f32(&VDt[iv][0], CHUNK, 0, lane);
      v16bf bb = ld_b_bt(&Mm[ic][0], CHUNK, 0, lane);
      oacc = wmma_bf(a, bb, oacc);
#pragma unroll
      for (int r = 0; r < 8; ++r) {
        int v = iv + r + mh, c = ic + nl;
        o[(size_t)(tokbase + t0 + c) * 2048 + vcol + v] = oacc[r];
      }
    }
    __syncthreads();

    // ---- state update: S^T += VD^T K  (32 subtiles over 8 waves) ----
    {
      int vt = (w & 1) * 16, kt0 = (w >> 1) * 64;
      v16bf a = ld_a_f32(&VDt[vt][0], CHUNK, 0, lane);
#pragma unroll
      for (int kt = kt0; kt < kt0 + 64; kt += 16) {
        v8f cacc;
#pragma unroll
        for (int r = 0; r < 8; ++r) cacc[r] = (float)Sbt[vt + r + mh][kt + nl];
        v16bf bb = ld_b_rm(&Kc[0][kt], DKH, lane);
        cacc = wmma_bf(a, bb, cacc);
#pragma unroll
        for (int r = 0; r < 8; ++r) Sbt[vt + r + mh][kt + nl] = (__bf16)cacc[r];
      }
    }
    __syncthreads();
  }
}

// ---------------- 6) per-head RMSNorm of o ---------------------------------

__global__ __launch_bounds__(256) void k_rms(const float* __restrict__ o,
                                             const float* __restrict__ onw,
                                             __bf16* __restrict__ ob) {
  __shared__ float red[256];
  int t = blockIdx.x, tid = threadIdx.x;
  const float* row = o + (size_t)t * 2048;
  float v[8]; float s = 0.f;
#pragma unroll
  for (int i = 0; i < 8; ++i) { v[i] = row[tid * 8 + i]; s += v[i] * v[i]; }
  red[tid] = s; __syncthreads();
  int g = tid >> 6, l = tid & 63;    // 4 groups of 64 threads: one 512-wide head each
  for (int st = 32; st; st >>= 1) { if (l < st) red[tid] += red[tid + st]; __syncthreads(); }
  float inv = rsqrtf(red[g << 6] * (1.f / DVH) + 1e-5f);
#pragma unroll
  for (int i = 0; i < 8; ++i) {
    int col = tid * 8 + i;
    ob[(size_t)t * 2048 + col] = (__bf16)(v[i] * inv * onw[col & (DVH - 1)]);
  }
}

// ---------------- host launch ----------------------------------------------

extern "C" void kernel_launch(void* const* d_in, const int* in_sizes, int n_in,
                              void* d_out, int out_size, void* d_ws, size_t ws_size,
                              hipStream_t stream) {
  (void)in_sizes; (void)n_in; (void)out_size; (void)ws_size;
  const float* x   = (const float*)d_in[0];
  const float* lnw = (const float*)d_in[1];
  const float* lnb = (const float*)d_in[2];
  const float* Wq  = (const float*)d_in[3];
  const float* Wk  = (const float*)d_in[4];
  const float* Wv  = (const float*)d_in[5];
  const float* Wb  = (const float*)d_in[6];
  const float* onw = (const float*)d_in[7];
  const float* Wo  = (const float*)d_in[8];
  float* out = (float*)d_out;

  char* p = (char*)d_ws;
  auto alloc = [&](size_t bytes) {
    char* r = p; p += (bytes + 255) & ~(size_t)255; return r;
  };
  __bf16* normed = (__bf16*)alloc((size_t)NTOK * DMODEL * 2);
  __bf16* wqkvt  = (__bf16*)alloc((size_t)4096 * 1024 * 2);
  __bf16* wot    = (__bf16*)alloc((size_t)1024 * 2048 * 2);
  float*  qk     = (float*) alloc((size_t)NTOK * 2048 * 4);
  __bf16* vbuf   = (__bf16*)alloc((size_t)NTOK * 2048 * 2);
  float*  betab  = (float*) alloc((size_t)NTOK * NH * 4);
  __bf16* qbuf   = (__bf16*)alloc((size_t)NTOK * 1024 * 2);
  __bf16* kbuf   = (__bf16*)alloc((size_t)NTOK * 1024 * 2);
  float*  obuf   = (float*) alloc((size_t)NTOK * 2048 * 4);
  __bf16* obbuf  = (__bf16*)alloc((size_t)NTOK * 2048 * 2);

  k_ln<<<NTOK, 256, 0, stream>>>(x, lnw, lnb, Wb, normed, betab);

  k_tr<<<(1024 * 1024 + 255) / 256, 256, 0, stream>>>(Wq, wqkvt,               1024, 1024);
  k_tr<<<(1024 * 1024 + 255) / 256, 256, 0, stream>>>(Wk, wqkvt + 1024 * 1024, 1024, 1024);
  k_tr<<<(1024 * 2048 + 255) / 256, 256, 0, stream>>>(Wv, wqkvt + 2048 * 1024, 1024, 2048);
  k_tr<<<(2048 * 1024 + 255) / 256, 256, 0, stream>>>(Wo, wot,                 2048, 1024);

  k_gemm_qkv<<<dim3(NTOK / 128, 4096 / 128), 256, 0, stream>>>(normed, wqkvt, qk, vbuf);

  k_l2n<<<NTOK, 256, 0, stream>>>(qk, qbuf, kbuf);

  k_scan<<<dim3(DVH / DVS, NB * NH), 256, 0, stream>>>(qbuf, kbuf, vbuf, betab, obuf);

  k_rms<<<NTOK, 256, 0, stream>>>(obuf, onw, obbuf);

  k_gemm_out<<<dim3(NTOK / 128, DMODEL / 128), 256, 0, stream>>>(obbuf, wot, x, out);
}
